// RNNSequence_11570641895535
// MI455X (gfx1250) — matrix operand order repaired and checked by reading, not compiled, not run on hardware
//
#include <hip/hip_runtime.h>

// RNN scan: out[b,t,:] = h_{t+1} = tanh(x[b,t,:]@Wx + b + h_t@Wh), h_0 = 0
// B=64, T=4096, D_IN=H=128.  fp32 in/out, bf16 WMMA with fp32 accumulate.
//
// Phase 1 (bandwidth-bound, whole device): xproj = x@Wx + b  -> d_ws (fp32)
// Phase 2 (latency-bound scan, 4 WGs):     h = tanh(wmma(h, Wh, C=xproj_t))

typedef __attribute__((ext_vector_type(16))) __bf16 v16bf;
typedef __attribute__((ext_vector_type(8)))  __bf16 v8bf;
typedef __attribute__((ext_vector_type(8)))  float  v8f;
typedef __attribute__((ext_vector_type(4)))  float  v4f;

constexpr int kB = 64;
constexpr int kT = 4096;
constexpr int kH = 128;   // == D_IN

// ---- native tanh (V_TANH_F32) or branchless TRANS-pipe fallback ----
__device__ __forceinline__ float fast_tanh(float v) {
#if __has_builtin(__builtin_amdgcn_tanhf)
    return __builtin_amdgcn_tanhf(v);
#else
    // tanh(v) = 1 - 2/(exp2(2*log2e*v) + 1); saturates to +-1 for large |v|.
    const float e = __builtin_amdgcn_exp2f(v * 2.8853900817779268f);
    return 1.0f - 2.0f * __builtin_amdgcn_rcpf(e + 1.0f);
#endif
}

// ---- A fragment (16x32 bf16) from a row-major [16][128] bf16 LDS buffer ----
// ISA 7.12.2: lanes 0-15 (M=lane): VGPR0-3 = K 0..7, VGPR4-7 = K 16..23;
//             lanes 16-31 (M=lane-16): VGPR0-3 = K 8..15, VGPR4-7 = K 24..31.
__device__ __forceinline__ v16bf load_a_frag(const __bf16* buf, int lane, int kc) {
    const int row  = lane & 15;
    const int half = lane >> 4;
    const int k0   = kc * 32 + half * 8;
    const int k1   = k0 + 16;
    union { v16bf v; v8bf h[2]; } u;
    u.h[0] = *(const v8bf*)(buf + row * kH + k0);   // ds_load_b128
    u.h[1] = *(const v8bf*)(buf + row * kH + k1);   // ds_load_b128
    return u.v;
}

// ---- B fragment (32x16 bf16) gathered from global fp32 W[k][n], once ----
__device__ __forceinline__ v16bf load_b_frag(const float* __restrict__ W,
                                             int lane, int n0, int kc) {
    const int n  = n0 + (lane & 15);
    const int kh = kc * 32 + ((lane >> 4) << 3);
    v16bf f;
#pragma unroll
    for (int i = 0; i < 8; ++i) {
        const int ko = (i >= 4) ? 8 : 0;
        f[2 * i]     = (__bf16)W[(kh + ko + 2 * i    ) * kH + n];
        f[2 * i + 1] = (__bf16)W[(kh + ko + 2 * i + 1) * kH + n];
    }
    return f;
}

#define WMMA_BF16(A, Bf, C) \
    __builtin_amdgcn_wmma_f32_16x16x32_bf16(false, (A), false, (Bf), (short)0, (C), false, false)

// =====================================================================
// Phase 1: xproj[m, :] = x[m, :] @ Wx + b   (m = b*T + t, 262144 rows)
// 2048 blocks x 256 threads; block covers 128 rows x 128 cols.
// =====================================================================
__global__ __launch_bounds__(256)
void xproj_kernel(const float* __restrict__ x, const float* __restrict__ Wx,
                  const float* __restrict__ bias, float* __restrict__ xp) {
    __shared__ __bf16 xs[128 * kH];   // 32 KB: 128-row bf16 tile of x

    const int tid  = threadIdx.x;
    const int lane = tid & 31;
    const int wave = tid >> 5;
    const int n0   = wave * 16;
    const size_t m0 = (size_t)blockIdx.x * 128;

    const int N     = lane & 15;
    const int Mbase = (lane >> 4) * 8;

    v16bf BF[4];
#pragma unroll
    for (int kc = 0; kc < 4; ++kc) BF[kc] = load_b_frag(Wx, lane, n0, kc);
    const float bv = bias[n0 + N];

    // stage 128x128 fp32 -> bf16 LDS (coalesced, streamed)
#pragma unroll
    for (int it = 0; it < 8; ++it) {
        const int idx = it * 256 + tid;        // 0..2047, 8 elems each
        const int row = idx >> 4;
        const int col = (idx & 15) * 8;
        const float* src = x + (m0 + row) * kH + col;
        v4f f0 = __builtin_nontemporal_load((const v4f*)src);
        v4f f1 = __builtin_nontemporal_load((const v4f*)src + 1);
        v8bf vv;
        vv[0] = (__bf16)f0[0]; vv[1] = (__bf16)f0[1];
        vv[2] = (__bf16)f0[2]; vv[3] = (__bf16)f0[3];
        vv[4] = (__bf16)f1[0]; vv[5] = (__bf16)f1[1];
        vv[6] = (__bf16)f1[2]; vv[7] = (__bf16)f1[3];
        *(v8bf*)(&xs[row * kH + col]) = vv;
    }
    __syncthreads();

#pragma unroll 2
    for (int rg = 0; rg < 8; ++rg) {
        const __bf16* xrow = xs + rg * 16 * kH;
        v16bf a0 = load_a_frag(xrow, lane, 0);
        v16bf a1 = load_a_frag(xrow, lane, 1);
        v16bf a2 = load_a_frag(xrow, lane, 2);
        v16bf a3 = load_a_frag(xrow, lane, 3);
        v8f z = {};
        v8f c0 = WMMA_BF16(a0, BF[0], z);
        v8f c1 = WMMA_BF16(a2, BF[2], z);
        c0 = WMMA_BF16(a1, BF[1], c0);
        c1 = WMMA_BF16(a3, BF[3], c1);
#pragma unroll
        for (int r = 0; r < 8; ++r)
            xp[(m0 + rg * 16 + Mbase + r) * kH + n0 + N] = c0[r] + c1[r] + bv;
    }
}

// =====================================================================
// Phase 2: the serial scan.  4 blocks x 256 threads (16 batch rows each,
// wave w owns h columns [16w, 16w+16)).  One barrier per timestep.
// =====================================================================
__device__ __forceinline__ void scan_step(
    int t, int lane, int n0, int b0, int N, int Mbase,
    const v16bf* WhF, const float* __restrict__ xp,
    __bf16* hb_cur, v8f& xp_cur, v8f& xp_next, v8f& h,
    float* __restrict__ out) {
    // publish h(t) as bf16 into this step's buffer
#pragma unroll
    for (int r = 0; r < 8; ++r)
        hb_cur[(Mbase + r) * kH + n0 + N] = (__bf16)h[r];

    // prefetch xproj(t+1) into registers (off the critical chain)
    const int tn = (t + 1 < kT) ? (t + 1) : t;
#pragma unroll
    for (int r = 0; r < 8; ++r)
        xp_next[r] = __builtin_nontemporal_load(
            xp + (((size_t)(b0 + Mbase + r) * kT) + tn) * kH + n0 + N);

    __syncthreads();   // single barrier per step (alternating h buffers)

    // all fragment loads issued before the first WMMA -> overlapped waits
    v16bf a0 = load_a_frag(hb_cur, lane, 0);
    v16bf a1 = load_a_frag(hb_cur, lane, 1);
    v16bf a2 = load_a_frag(hb_cur, lane, 2);
    v16bf a3 = load_a_frag(hb_cur, lane, 3);
    v8f z  = {};
    v8f c0 = WMMA_BF16(a0, WhF[0], xp_cur);   // C = xproj(t): no extra add
    v8f c1 = WMMA_BF16(a2, WhF[2], z);        // two independent 2-deep chains
    c0 = WMMA_BF16(a1, WhF[1], c0);
    c1 = WMMA_BF16(a3, WhF[3], c1);

#pragma unroll
    for (int j = 0; j < 8; ++j)
        h[j] = fast_tanh(c0[j] + c1[j]);

#pragma unroll
    for (int r = 0; r < 8; ++r)
        __builtin_nontemporal_store(h[r],
            out + (((size_t)(b0 + Mbase + r) * kT) + t) * kH + n0 + N);
}

__global__ __launch_bounds__(256)
void rnn_scan(const float* __restrict__ xp, const float* __restrict__ Wh,
              float* __restrict__ out) {
    __shared__ __bf16 hbuf0[16 * kH];
    __shared__ __bf16 hbuf1[16 * kH];

    const int tid  = threadIdx.x;
    const int lane = tid & 31;
    const int wave = tid >> 5;
    const int n0   = wave * 16;
    const int b0   = blockIdx.x * 16;
    const int N     = lane & 15;
    const int Mbase = (lane >> 4) * 8;

    v16bf WhF[4];
#pragma unroll
    for (int kc = 0; kc < 4; ++kc) WhF[kc] = load_b_frag(Wh, lane, n0, kc);

    v8f h = {};                     // h_0 = 0
    v8f xpA, xpB;
#pragma unroll
    for (int r = 0; r < 8; ++r)     // preload xproj(0)
        xpA[r] = __builtin_nontemporal_load(
            xp + ((size_t)(b0 + Mbase + r) * kT) * kH + n0 + N);

    for (int t = 0; t < kT; t += 2) {   // x2 unroll: static reg double-buffer
        scan_step(t,     lane, n0, b0, N, Mbase, WhF, xp, hbuf0, xpA, xpB, h, out);
        scan_step(t + 1, lane, n0, b0, N, Mbase, WhF, xp, hbuf1, xpB, xpA, h, out);
    }
}

// =====================================================================
// Fallback: fused single-kernel scan (if workspace too small for xproj).
// =====================================================================
__global__ __launch_bounds__(256)
void rnn_scan_fused(const float* __restrict__ x, const float* __restrict__ Wx,
                    const float* __restrict__ Wh, const float* __restrict__ bias,
                    float* __restrict__ out) {
    __shared__ __bf16 xbuf[2][16 * kH];
    __shared__ __bf16 hbuf[2][16 * kH];

    const int tid  = threadIdx.x;
    const int lane = tid & 31;
    const int wave = tid >> 5;
    const int n0   = wave * 16;
    const int b0   = blockIdx.x * 16;
    const int N     = lane & 15;
    const int Mbase = (lane >> 4) * 8;

    v16bf WxF[4], WhF[4];
#pragma unroll
    for (int kc = 0; kc < 4; ++kc) {
        WxF[kc] = load_b_frag(Wx, lane, n0, kc);
        WhF[kc] = load_b_frag(Wh, lane, n0, kc);
    }
    const float bv = bias[n0 + N];
    const int srow = tid >> 4;
    const int scol = (tid & 15) * 8;

    v8f h = {};
    for (int t = 0; t < kT; ++t) {
        const int p = t & 1;
        {
            const float* src = x + (((size_t)(b0 + srow) * kT) + t) * kH + scol;
            v4f f0 = __builtin_nontemporal_load((const v4f*)src);
            v4f f1 = __builtin_nontemporal_load((const v4f*)src + 1);
            v8bf vv;
            vv[0] = (__bf16)f0[0]; vv[1] = (__bf16)f0[1];
            vv[2] = (__bf16)f0[2]; vv[3] = (__bf16)f0[3];
            vv[4] = (__bf16)f1[0]; vv[5] = (__bf16)f1[1];
            vv[6] = (__bf16)f1[2]; vv[7] = (__bf16)f1[3];
            *(v8bf*)(&xbuf[p][srow * kH + scol]) = vv;
        }
#pragma unroll
        for (int r = 0; r < 8; ++r)
            hbuf[p][(Mbase + r) * kH + n0 + N] = (__bf16)h[r];

        __syncthreads();

        v16bf ax0 = load_a_frag(xbuf[p], lane, 0);
        v16bf ax1 = load_a_frag(xbuf[p], lane, 1);
        v16bf ax2 = load_a_frag(xbuf[p], lane, 2);
        v16bf ax3 = load_a_frag(xbuf[p], lane, 3);
        v16bf ah0 = load_a_frag(hbuf[p], lane, 0);
        v16bf ah1 = load_a_frag(hbuf[p], lane, 1);
        v16bf ah2 = load_a_frag(hbuf[p], lane, 2);
        v16bf ah3 = load_a_frag(hbuf[p], lane, 3);
        v8f z = {};
        v8f c0 = WMMA_BF16(ax0, WxF[0], z);
        v8f c1 = WMMA_BF16(ax2, WxF[2], z);
        c0 = WMMA_BF16(ax1, WxF[1], c0);
        c1 = WMMA_BF16(ax3, WxF[3], c1);
        c0 = WMMA_BF16(ah0, WhF[0], c0);
        c1 = WMMA_BF16(ah2, WhF[2], c1);
        c0 = WMMA_BF16(ah1, WhF[1], c0);
        c1 = WMMA_BF16(ah3, WhF[3], c1);

#pragma unroll
        for (int j = 0; j < 8; ++j)
            h[j] = fast_tanh(c0[j] + c1[j] + bv);

#pragma unroll
        for (int r = 0; r < 8; ++r)
            __builtin_nontemporal_store(h[r],
                out + (((size_t)(b0 + Mbase + r) * kT) + t) * kH + n0 + N);
    }
}

extern "C" void kernel_launch(void* const* d_in, const int* in_sizes, int n_in,
                              void* d_out, int out_size, void* d_ws, size_t ws_size,
                              hipStream_t stream) {
    (void)in_sizes; (void)n_in; (void)out_size;
    const float* x  = (const float*)d_in[0];   // [B, T, D_IN] fp32
    const float* Wx = (const float*)d_in[1];   // [D_IN, H]    fp32
    const float* Wh = (const float*)d_in[2];   // [H, H]       fp32
    const float* b  = (const float*)d_in[3];   // [H]          fp32
    float* out = (float*)d_out;                // [B, T, H]    fp32

    const size_t need = (size_t)kB * kT * kH * sizeof(float);  // 134 MB
    if (ws_size >= need) {
        float* xp = (float*)d_ws;
        xproj_kernel<<<(kB * kT) / 128, 256, 0, stream>>>(x, Wx, b, xp);
        rnn_scan<<<kB / 16, 256, 0, stream>>>(xp, Wh, out);
    } else {
        rnn_scan_fused<<<kB / 16, 256, 0, stream>>>(x, Wx, Wh, b, out);
    }
}
